// TransformerEncoder_88210038326402
// MI455X (gfx1250) — matrix operand order repaired and checked by reading, not compiled, not run on hardware
//
#include <hip/hip_runtime.h>
#include <math.h>

// ---------------------------------------------------------------------------
// Transformer encoder block for MI455X (gfx1250), wave32 + WMMA bf16 + TDM.
// GEMMs: V_WMMA_F32_16X16X32_BF16 with double-buffered TENSOR_LOAD_TO_LDS.
// Attention: flash-style streaming softmax (no 384MB score tensor).
// ---------------------------------------------------------------------------

typedef __attribute__((ext_vector_type(16))) __bf16   v16bf_t;
typedef __attribute__((ext_vector_type(8)))  float    v8f_t;
typedef __attribute__((ext_vector_type(8)))  unsigned v8u_t;
typedef __attribute__((ext_vector_type(4)))  unsigned u32x4_t;
typedef __attribute__((ext_vector_type(4)))  int      i32x4_t;
typedef __attribute__((ext_vector_type(8)))  int      i32x8_t;

__device__ __forceinline__ unsigned short f2bf(float f) {
    unsigned u = __builtin_bit_cast(unsigned, f);
    u += 0x7FFFu + ((u >> 16) & 1u);   // round-to-nearest-even
    return (unsigned short)(u >> 16);
}

__device__ __forceinline__ v8f_t wmma_bf16(v8u_t a, v8u_t b, v8f_t c) {
    return __builtin_amdgcn_wmma_f32_16x16x32_bf16(
        false, __builtin_bit_cast(v16bf_t, a),
        false, __builtin_bit_cast(v16bf_t, b),
        (short)0, c, false, false);
}

// A-fragment (16x32 bf16): lane m=lane%16, pair k0 = 16*(j/4)+2*(j%4)+8*(lane/16)
__device__ __forceinline__ v8u_t load_frag_a(const unsigned short* base, int stride, int lane) {
    const unsigned short* p = base + (lane & 15) * stride;
    const int half8 = (lane >> 4) << 3;
    v8u_t r;
#pragma unroll
    for (int j = 0; j < 8; ++j) {
        int k0 = ((j >> 2) << 4) + ((j & 3) << 1) + half8;
        r[j] = *(const unsigned*)(p + k0);
    }
    return r;
}

// B-fragment (32x16 bf16), LDS stored [n][k]: lane n=lane%16, pair k0 = 16*(j/4)+2*(j%4)+8*(lane/16)
__device__ __forceinline__ v8u_t load_frag_b(const unsigned short* base, int stride, int lane) {
    const unsigned short* p = base + (lane & 15) * stride;
    const int half8 = (lane >> 4) << 3;
    v8u_t r;
#pragma unroll
    for (int j = 0; j < 8; ++j) {
        int k0 = ((j >> 2) << 4) + ((j & 3) << 1) + half8;
        r[j] = *(const unsigned*)(p + k0);
    }
    return r;
}

// ---------------------------------------------------------------------------
// TDM: 2D tile (tile_d0 contiguous elems x tile_d1 rows), 2-byte elements,
// LDS padding pad_amount DW after every pad_interval DW. D# per ISA 8.3/8.4.
// ---------------------------------------------------------------------------
__device__ __forceinline__ void tdm_load_tile_2d_bf16(
    unsigned lds_addr, const unsigned short* gptr,
    unsigned tile_d0, unsigned tile_d1,
    unsigned tensor_d0, unsigned tensor_d1,
    unsigned long long stride0_elems,
    unsigned pad_interval_code, unsigned pad_amount_code) {
    unsigned long long ga = (unsigned long long)(uintptr_t)gptr;
    u32x4_t g0;
    g0[0] = 1u;                                   // count=1, user mode, no gather
    g0[1] = lds_addr;                             // LDS byte address
    g0[2] = (unsigned)ga;                         // global_addr[31:0]
    g0[3] = (unsigned)(ga >> 32) | (2u << 30);    // global_addr[56:32] | type=2
    i32x8_t g1;
    g1[0] = (int)((1u << 16)                      // data_size = 2 bytes
                | (1u << 20)                      // pad_enable
                | (pad_interval_code << 22)
                | (pad_amount_code << 25));
    g1[1] = (int)((tensor_d0 & 0xFFFFu) << 16);   // tensor_dim0[15:0] @ bits 63:48
    g1[2] = (int)((tensor_d0 >> 16) | ((tensor_d1 & 0xFFFFu) << 16));
    g1[3] = (int)((tensor_d1 >> 16) | (tile_d0 << 16));
    g1[4] = (int)(tile_d1 & 0xFFFFu);             // tile_dim1 (tile_dim2 = 0)
    g1[5] = (int)(unsigned)stride0_elems;         // tensor_dim0_stride[31:0]
    g1[6] = (int)(unsigned)((stride0_elems >> 32) & 0xFFFFu);
    g1[7] = 0;
    i32x4_t gz = {0, 0, 0, 0};
#if __clang_major__ >= 23
    i32x8_t gz8 = {};
    __builtin_amdgcn_tensor_load_to_lds(g0, g1, gz, gz, gz8, 0);
#else
    __builtin_amdgcn_tensor_load_to_lds(g0, g1, gz, gz, 0);
#endif
}

// ---------------------------------------------------------------------------
// Weight convert + transpose: W[K][N] f32 -> Wt[N][K] bf16
// ---------------------------------------------------------------------------
__global__ __launch_bounds__(256)
void te_transpose_bf16(const float* __restrict__ W, unsigned short* __restrict__ Wt,
                       int K, int N) {
    int id = blockIdx.x * 256 + threadIdx.x;
    if (id >= K * N) return;
    int k = id / N, n = id - k * N;           // coalesced read along N
    Wt[(size_t)n * K + k] = f2bf(W[id]);
}

// ---------------------------------------------------------------------------
// LayerNorm (C = 768) f32 -> bf16, one block per row
// ---------------------------------------------------------------------------
__global__ __launch_bounds__(256)
void te_layernorm_bf16(const float* __restrict__ x, const float* __restrict__ g,
                       const float* __restrict__ b, unsigned short* __restrict__ out) {
    const int row = blockIdx.x, tid = threadIdx.x;
    const float* xr = x + (size_t)row * 768;
    float a0 = xr[tid], a1 = xr[tid + 256], a2 = xr[tid + 512];
    __shared__ float sbuf[256];
    sbuf[tid] = a0 + a1 + a2;
    __syncthreads();
    for (int s = 128; s > 0; s >>= 1) { if (tid < s) sbuf[tid] += sbuf[tid + s]; __syncthreads(); }
    float mean = sbuf[0] * (1.0f / 768.0f);
    __syncthreads();
    float d0 = a0 - mean, d1 = a1 - mean, d2 = a2 - mean;
    sbuf[tid] = d0 * d0 + d1 * d1 + d2 * d2;
    __syncthreads();
    for (int s = 128; s > 0; s >>= 1) { if (tid < s) sbuf[tid] += sbuf[tid + s]; __syncthreads(); }
    float rstd = rsqrtf(sbuf[0] * (1.0f / 768.0f) + 1e-5f);
    unsigned short* o = out + (size_t)row * 768;
    o[tid]       = f2bf(d0 * rstd * g[tid]       + b[tid]);
    o[tid + 256] = f2bf(d1 * rstd * g[tid + 256] + b[tid + 256]);
    o[tid + 512] = f2bf(d2 * rstd * g[tid + 512] + b[tid + 512]);
}

// ---------------------------------------------------------------------------
// Tiled bf16 GEMM: C[M][N] = A[M][K] * Bt[N][K]^T + bias, fused epilogues.
// Block 256 thr = 8 waves (2x4), tile 128x128x32, per-wave 64x32 (8 WMMA accs).
// Global->LDS via double-buffered TDM (wave 0 issues, TENSORcnt + barrier).
// ---------------------------------------------------------------------------
#define EPI_BF16  0
#define EPI_RESID 1
#define EPI_GELU  2
#define EPI_QKV   3

template <int EPI>
__global__ __launch_bounds__(256)
void te_gemm_wmma(const unsigned short* __restrict__ A,
                  const unsigned short* __restrict__ Bt,
                  const float* __restrict__ bias,
                  const float* __restrict__ resid,
                  float* __restrict__ outF,
                  unsigned short* __restrict__ outH,
                  float* __restrict__ qkvF,
                  unsigned short* __restrict__ qbh,
                  unsigned short* __restrict__ kbh,
                  unsigned short* __restrict__ vt,
                  int M, int N, int K) {
    constexpr int BM = 128, BN = 128, BK = 32, SK = BK + 8;  // 40-half rows
    // TDM pad: tile row = 32 halves = 16 DW (interval code 3), pad 4 DW (code 3)
    __shared__ __align__(16) unsigned short As[2][BM * SK];
    __shared__ __align__(16) unsigned short Bs[2][BN * SK];

    const int tid = threadIdx.x;
    const int lane = tid & 31, wid = tid >> 5, half = lane >> 4;
    const int wm = (wid >> 2) * 64, wn = (wid & 3) * 32;
    const int m0 = blockIdx.y * BM, n0 = blockIdx.x * BN;
    const int niter = K / BK;

    const unsigned ldsA[2] = { (unsigned)(uintptr_t)&As[0][0], (unsigned)(uintptr_t)&As[1][0] };
    const unsigned ldsB[2] = { (unsigned)(uintptr_t)&Bs[0][0], (unsigned)(uintptr_t)&Bs[1][0] };
    const unsigned short* Abase = A  + (size_t)m0 * K;
    const unsigned short* Bbase = Bt + (size_t)n0 * K;

    v8f_t acc[4][2] = {};

    if (wid == 0) {  // prologue: DMA k-step 0 into buffer 0
        tdm_load_tile_2d_bf16(ldsA[0], Abase, BK, BM, (unsigned)K, (unsigned)M,
                              (unsigned long long)K, 3u, 3u);
        tdm_load_tile_2d_bf16(ldsB[0], Bbase, BK, BN, (unsigned)K, (unsigned)N,
                              (unsigned long long)K, 3u, 3u);
    }

    for (int i = 0; i < niter; ++i) {
        const int buf = i & 1;
        __syncthreads();           // previous compute done: next buffer reusable
        if (wid == 0) {
            if (i + 1 < niter) {   // DMA k-step i+1 while we compute step i
                tdm_load_tile_2d_bf16(ldsA[buf ^ 1], Abase + (i + 1) * BK, BK, BM,
                                      (unsigned)K, (unsigned)M, (unsigned long long)K, 3u, 3u);
                tdm_load_tile_2d_bf16(ldsB[buf ^ 1], Bbase + (i + 1) * BK, BK, BN,
                                      (unsigned)K, (unsigned)N, (unsigned long long)K, 3u, 3u);
                __builtin_amdgcn_s_wait_tensorcnt(2);  // step-i pair complete (in-order)
            } else {
                __builtin_amdgcn_s_wait_tensorcnt(0);
            }
        }
        __syncthreads();           // step-i tiles visible to all waves

        v8u_t av[4], bv[2];
#pragma unroll
        for (int mi = 0; mi < 4; ++mi)
            av[mi] = load_frag_a(&As[buf][(wm + mi * 16) * SK], SK, lane);
#pragma unroll
        for (int ni = 0; ni < 2; ++ni)
            bv[ni] = load_frag_b(&Bs[buf][(wn + ni * 16) * SK], SK, lane);
#pragma unroll
        for (int mi = 0; mi < 4; ++mi)
#pragma unroll
            for (int ni = 0; ni < 2; ++ni)
                acc[mi][ni] = wmma_bf16(av[mi], bv[ni], acc[mi][ni]);
    }

    // Epilogue. C/D layout: row = r + 8*(lane/16), col = lane%16.
#pragma unroll
    for (int mi = 0; mi < 4; ++mi) {
#pragma unroll
        for (int ni = 0; ni < 2; ++ni) {
#pragma unroll
            for (int r = 0; r < 8; ++r) {
                int row = m0 + wm + mi * 16 + (half << 3) + r;
                int col = n0 + wn + ni * 16 + (lane & 15);
                float val = acc[mi][ni][r] + bias[col];
                if (EPI == EPI_RESID) {
                    outF[(size_t)row * N + col] = resid[(size_t)row * N + col] + val;
                } else if (EPI == EPI_GELU) {
                    float gl = 0.5f * val * (1.0f + erff(val * 0.70710678118654752f));
                    outH[(size_t)row * N + col] = f2bf(gl);
                } else if (EPI == EPI_QKV) {
                    int s = col / 768, rem = col - s * 768;
                    int h = rem >> 6, d = rem & 63;
                    int b = row >> 10, n = row & 1023;
                    int bhh = b * 12 + h;
                    // d_out qkv section: [3, B, H, N, Dh] f32
                    qkvF[((((size_t)s * 8 + b) * 12 + h) * 1024 + n) * 64 + d] = val;
                    unsigned short hv = f2bf(val);
                    size_t nd = ((size_t)bhh * 1024 + n) * 64 + d;
                    if (s == 0)      qbh[nd] = hv;                              // [bh][n][d]
                    else if (s == 1) kbh[nd] = hv;                              // [bh][n][d]
                    else             vt[((size_t)bhh * 64 + d) * 1024 + n] = hv; // [bh][d][n]
                } else {
                    outH[(size_t)row * N + col] = f2bf(val);
                }
            }
        }
    }
}

// ---------------------------------------------------------------------------
// Flash attention: grid (B*H=96, Nq/128=8), 256 thr = 8 waves.
// Each wave owns a 16-query strip; streams 64-key chunks with online softmax.
// ---------------------------------------------------------------------------
__global__ __launch_bounds__(256)
void te_attention_wmma(const unsigned short* __restrict__ qbh,
                       const unsigned short* __restrict__ kbh,
                       const unsigned short* __restrict__ vt,
                       unsigned short* __restrict__ obf) {
    constexpr int SD = 72;  // 64 + 8 pad (halves)
    __shared__ __align__(16) unsigned short Qs[128 * SD];      // [q][d]
    __shared__ __align__(16) unsigned short Ks[64 * SD];       // [key][d]
    __shared__ __align__(16) unsigned short Vs[64 * SD];       // [d][key] (from vt)
    __shared__ __align__(16) unsigned short Ps[8 * 16 * SD];   // per-wave P tile [m][key]

    const int bh = blockIdx.x;
    const int q0 = blockIdx.y * 128;
    const int tid = threadIdx.x, lane = tid & 31, wid = tid >> 5, half = lane >> 4;

    const size_t qbase = ((size_t)bh * 1024 + q0) * 64;
#pragma unroll
    for (int i = 0; i < 4; ++i) {   // 1024 x 16B chunks
        int c = tid + i * 256;
        int row = c >> 3, kc = (c & 7) << 3;
        *(uint4*)&Qs[row * SD + kc] = *(const uint4*)(qbh + qbase + (size_t)row * 64 + kc);
    }

    float mrow[8], lrow[8];
#pragma unroll
    for (int r = 0; r < 8; ++r) { mrow[r] = -1e30f; lrow[r] = 0.0f; }
    v8f_t oacc[4] = {};

    for (int kc0 = 0; kc0 < 1024; kc0 += 64) {
        __syncthreads();
#pragma unroll
        for (int i = 0; i < 2; ++i) {  // 512 x 16B chunks (K and V tiles)
            int c = tid + i * 256;
            int row = c >> 3, kcc = (c & 7) << 3;
            *(uint4*)&Ks[row * SD + kcc] =
                *(const uint4*)(kbh + ((size_t)bh * 1024 + kc0 + row) * 64 + kcc);
            *(uint4*)&Vs[row * SD + kcc] =
                *(const uint4*)(vt + ((size_t)bh * 64 + row) * 1024 + kc0 + kcc);
        }
        __syncthreads();

        // S = Q * K^T  (16 queries x 64 keys), K-dim = 64 -> 2 WMMA k-steps
        v8f_t sacc[4] = {};
#pragma unroll
        for (int kk = 0; kk < 2; ++kk) {
            v8u_t af = load_frag_a(&Qs[(wid * 16) * SD + kk * 32], SD, lane);
#pragma unroll
            for (int ni = 0; ni < 4; ++ni) {
                v8u_t bf_ = load_frag_b(&Ks[(ni * 16) * SD + kk * 32], SD, lane);
                sacc[ni] = wmma_bf16(af, bf_, sacc[ni]);
            }
        }

        // Online softmax. Row r+8*half lives in one 16-lane half -> shfl_xor {1,2,4,8}.
#pragma unroll
        for (int r = 0; r < 8; ++r) {
            float mx = -1e30f;
#pragma unroll
            for (int ni = 0; ni < 4; ++ni) {
                sacc[ni][r] *= 0.125f;   // 1/sqrt(64)
                mx = fmaxf(mx, sacc[ni][r]);
            }
#pragma unroll
            for (int off = 8; off >= 1; off >>= 1) mx = fmaxf(mx, __shfl_xor(mx, off, 32));
            float mnew = fmaxf(mrow[r], mx);
            float alpha = __expf(mrow[r] - mnew);
            mrow[r] = mnew;
            float sm = 0.0f;
            unsigned short* pr = &Ps[(wid * 16 + (half << 3) + r) * SD];
#pragma unroll
            for (int ni = 0; ni < 4; ++ni) {
                float pv = __expf(sacc[ni][r] - mnew);
                sm += pv;
                pr[ni * 16 + (lane & 15)] = f2bf(pv);
            }
#pragma unroll
            for (int off = 8; off >= 1; off >>= 1) sm += __shfl_xor(sm, off, 32);
            lrow[r] = lrow[r] * alpha + sm;
#pragma unroll
            for (int ni = 0; ni < 4; ++ni) oacc[ni][r] *= alpha;
        }
        __builtin_amdgcn_sched_barrier(0);  // keep P stores before P loads

        // O += P * V  (K-dim = 64 keys -> 2 WMMA k-steps, N-dim = 64 dims)
#pragma unroll
        for (int kk = 0; kk < 2; ++kk) {
            v8u_t af = load_frag_a(&Ps[(wid * 16) * SD + kk * 32], SD, lane);
#pragma unroll
            for (int ni = 0; ni < 4; ++ni) {
                v8u_t bf_ = load_frag_b(&Vs[(ni * 16) * SD + kk * 32], SD, lane);
                oacc[ni] = wmma_bf16(af, bf_, oacc[ni]);
            }
        }
    }

    // Write O into [B, N, C] bf16 (C = h*64 + d) for the projection GEMM.
    const int b = bh / 12, h = bh % 12;
#pragma unroll
    for (int ni = 0; ni < 4; ++ni) {
#pragma unroll
        for (int r = 0; r < 8; ++r) {
            int q = q0 + wid * 16 + (half << 3) + r;
            int d = ni * 16 + (lane & 15);
            float val = oacc[ni][r] / lrow[r];
            obf[((size_t)b * 1024 + q) * 768 + h * 64 + d] = f2bf(val);
        }
    }
}

// ---------------------------------------------------------------------------
// Host launcher
// ---------------------------------------------------------------------------
extern "C" void kernel_launch(void* const* d_in, const int* in_sizes, int n_in,
                              void* d_out, int out_size, void* d_ws, size_t ws_size,
                              hipStream_t stream) {
    (void)in_sizes; (void)n_in; (void)out_size; (void)ws_size;
    const float* x      = (const float*)d_in[0];
    const float* W_qkv  = (const float*)d_in[1];
    const float* b_qkv  = (const float*)d_in[2];
    const float* W_proj = (const float*)d_in[3];
    const float* b_proj = (const float*)d_in[4];
    const float* g1     = (const float*)d_in[5];
    const float* be1    = (const float*)d_in[6];
    const float* g2     = (const float*)d_in[7];
    const float* be2    = (const float*)d_in[8];
    const float* W_fc1  = (const float*)d_in[9];
    const float* b_fc1  = (const float*)d_in[10];
    const float* W_fc2  = (const float*)d_in[11];
    const float* b_fc2  = (const float*)d_in[12];

    const int M = 8 * 1024;            // 8192 tokens
    float* out_x   = (float*)d_out;                     // [B,N,768]
    float* out_qkv = out_x + (size_t)M * 768;           // [3,B,H,N,64]

    // Workspace carve (256B aligned)
    char* p = (char*)d_ws;
    auto alloc = [&](size_t bytes) -> void* {
        void* r = (void*)p;
        p += (bytes + 255) & ~(size_t)255;
        return r;
    };
    unsigned short* WqkvT = (unsigned short*)alloc((size_t)2304 * 768 * 2);
    unsigned short* WprjT = (unsigned short*)alloc((size_t)768 * 768 * 2);
    unsigned short* Wfc1T = (unsigned short*)alloc((size_t)3072 * 768 * 2);
    unsigned short* Wfc2T = (unsigned short*)alloc((size_t)768 * 3072 * 2);
    unsigned short* xn1   = (unsigned short*)alloc((size_t)M * 768 * 2);
    unsigned short* qbh   = (unsigned short*)alloc((size_t)96 * 1024 * 64 * 2);
    unsigned short* kbh   = (unsigned short*)alloc((size_t)96 * 1024 * 64 * 2);
    unsigned short* vtb   = (unsigned short*)alloc((size_t)96 * 64 * 1024 * 2);
    unsigned short* obf   = (unsigned short*)alloc((size_t)M * 768 * 2);
    float*          x1    = (float*)alloc((size_t)M * 768 * 4);
    unsigned short* xn2   = (unsigned short*)alloc((size_t)M * 768 * 2);
    unsigned short* hid   = (unsigned short*)alloc((size_t)M * 3072 * 2);

    // 1) weight convert + transpose to bf16 [N][K]
    te_transpose_bf16<<<(768 * 2304 + 255) / 256, 256, 0, stream>>>(W_qkv, WqkvT, 768, 2304);
    te_transpose_bf16<<<(768 * 768  + 255) / 256, 256, 0, stream>>>(W_proj, WprjT, 768, 768);
    te_transpose_bf16<<<(768 * 3072 + 255) / 256, 256, 0, stream>>>(W_fc1, Wfc1T, 768, 3072);
    te_transpose_bf16<<<(3072 * 768 + 255) / 256, 256, 0, stream>>>(W_fc2, Wfc2T, 3072, 768);

    // 2) LN1 -> bf16
    te_layernorm_bf16<<<M, 256, 0, stream>>>(x, g1, be1, xn1);

    // 3) QKV GEMM (8192 x 2304 x 768), scatter qkv f32 + per-head bf16 q/k/v^T
    te_gemm_wmma<EPI_QKV><<<dim3(2304 / 128, M / 128), 256, 0, stream>>>(
        xn1, WqkvT, b_qkv, nullptr, nullptr, nullptr, out_qkv, qbh, kbh, vtb, M, 2304, 768);

    // 4) flash attention per (b,h), 128-query tiles
    te_attention_wmma<<<dim3(96, 8), 256, 0, stream>>>(qbh, kbh, vtb, obf);

    // 5) proj GEMM + residual: x1 = x + o @ W_proj + b
    te_gemm_wmma<EPI_RESID><<<dim3(768 / 128, M / 128), 256, 0, stream>>>(
        obf, WprjT, b_proj, x, x1, nullptr, nullptr, nullptr, nullptr, nullptr, M, 768, 768);

    // 6) LN2 -> bf16
    te_layernorm_bf16<<<M, 256, 0, stream>>>(x1, g2, be2, xn2);

    // 7) fc1 GEMM + exact GELU -> bf16 hid
    te_gemm_wmma<EPI_GELU><<<dim3(3072 / 128, M / 128), 256, 0, stream>>>(
        xn2, Wfc1T, b_fc1, nullptr, nullptr, hid, nullptr, nullptr, nullptr, nullptr, M, 3072, 768);

    // 8) fc2 GEMM + residual -> d_out x section
    te_gemm_wmma<EPI_RESID><<<dim3(768 / 128, M / 128), 256, 0, stream>>>(
        hid, Wfc2T, b_fc2, x1, out_x, nullptr, nullptr, nullptr, nullptr, nullptr, M, 768, 3072);
}
